// S_CLSTM_DANN_33079838114752
// MI455X (gfx1250) — 8-bit matrix operand order repaired and checked by reading, compiled, not run on hardware
//
#include <hip/hip_runtime.h>
#include <hip/hip_bf16.h>

#define T_LEN 2000
#define BATCH 256
#define CIN   14
#define CH    64
#define HID   128
#define G4    512
#define K1    192
#define K2    256

typedef __attribute__((ext_vector_type(8))) int   v8i_t;
typedef __attribute__((ext_vector_type(8))) float v8f_t;

__device__ __forceinline__ float sigm(float x) { return 1.0f / (1.0f + __expf(-x)); }

__device__ __forceinline__ float fast_tanh(float x) {
    float e = __expf(2.0f * x);
    return 1.0f - 2.0f / (e + 1.0f);
}

__device__ __forceinline__ unsigned f32_to_e4m3(float f) {
    unsigned u = __float_as_uint(f);
    unsigned s = (u >> 24) & 0x80u;
    int e = (int)((u >> 23) & 0xffu) - 127;
    unsigned m = (u >> 20) & 0x7u;
    if (((u >> 23) & 0xffu) == 0u) return s;
    if (e < -6) return s;
    if (e > 8)  return s | 0x7Eu;
    return s | (unsigned)((e + 7) << 3) | m;
}

__global__ void zero_stats_kernel(float* stats) {
    if (threadIdx.x < 128) stats[threadIdx.x] = 0.0f;
}

__global__ __launch_bounds__(256) void conv_stats_kernel(const float* __restrict__ x,
                                                         const float* __restrict__ w,
                                                         __hip_bfloat16* __restrict__ convb,
                                                         float* __restrict__ stats) {
    __shared__ float wl[CH * CIN * 5];
    __shared__ float redS[256];
    __shared__ float redQ[256];
    for (int i = threadIdx.x; i < CH * CIN * 5; i += blockDim.x) wl[i] = w[i];
    __syncthreads();

    const int total  = T_LEN * BATCH * CH;
    const int stride = gridDim.x * blockDim.x;
    float ls = 0.0f, lq = 0.0f;
    const int ch = threadIdx.x & 63;
    const float* wc = &wl[ch * CIN * 5];

    for (int idx = blockIdx.x * blockDim.x + threadIdx.x; idx < total; idx += stride) {
        int tb = idx >> 6;
        int b  = tb & (BATCH - 1);
        int t  = tb >> 8;
        float acc = 0.0f;
#pragma unroll
        for (int k = 0; k < 5; ++k) {
            int tp = t + k - 2;
            if (tp < 0 || tp >= T_LEN) continue;
            const float* xr = &x[(tp * BATCH + b) * CIN];
#pragma unroll
            for (int c = 0; c < CIN; ++c) acc += xr[c] * wc[c * 5 + k];
        }
        convb[idx] = __float2bfloat16(acc);
        ls += acc;
        lq += acc * acc;
    }
    redS[threadIdx.x] = ls;
    redQ[threadIdx.x] = lq;
    __syncthreads();
    if (threadIdx.x < 64) {
        float s = redS[threadIdx.x] + redS[threadIdx.x + 64] + redS[threadIdx.x + 128] + redS[threadIdx.x + 192];
        float q = redQ[threadIdx.x] + redQ[threadIdx.x + 64] + redQ[threadIdx.x + 128] + redQ[threadIdx.x + 192];
        atomicAdd(&stats[threadIdx.x], s);
        atomicAdd(&stats[64 + threadIdx.x], q);
    }
}

__global__ void bn_finalize_kernel(const float* __restrict__ stats,
                                   const float* __restrict__ gamma,
                                   const float* __restrict__ beta,
                                   const float* __restrict__ thr_lif1,
                                   float* __restrict__ scale,
                                   float* __restrict__ shiftT) {
    int c = threadIdx.x;
    if (c >= 64) return;
    const float invN = 1.0f / (float)(T_LEN * BATCH);
    float mean = stats[c] * invN;
    float var  = stats[64 + c] * invN - mean * mean;
    float sc   = gamma[c] * rsqrtf(var + 1e-5f);
    scale[c]  = sc;
    shiftT[c] = beta[c] - mean * sc - thr_lif1[0];
}

__global__ void pack_kernel(const float* __restrict__ wih1, const float* __restrict__ whh1,
                            const float* __restrict__ bih1, const float* __restrict__ bhh1,
                            const float* __restrict__ wih2, const float* __restrict__ whh2,
                            const float* __restrict__ bih2, const float* __restrict__ bhh2,
                            unsigned char* __restrict__ W1, unsigned char* __restrict__ W2,
                            float* __restrict__ bias1, float* __restrict__ bias2) {
    const int stride = gridDim.x * blockDim.x;
    const int tid0 = blockIdx.x * blockDim.x + threadIdx.x;
    for (int i = tid0; i < G4 * K1; i += stride) {
        int n = i / K1, k = i - n * K1;
        float v = (k < CH) ? wih1[n * CH + k] : whh1[n * HID + (k - CH)];
        W1[i] = (unsigned char)f32_to_e4m3(v);
    }
    for (int i = tid0; i < G4 * K2; i += stride) {
        int n = i / K2, k = i - n * K2;
        float v = (k < HID) ? wih2[n * HID + k] : whh2[n * HID + (k - HID)];
        W2[i] = (unsigned char)f32_to_e4m3(v);
    }
    for (int i = tid0; i < G4; i += stride) {
        bias1[i] = bih1[i] + bhh1[i];
        bias2[i] = bih2[i] + bhh2[i];
    }
}

#define LDS_W1    0
#define LDS_W2    98304
#define LDS_A1    229376
#define LDS_A2    232448
#define LDS_GATES 236544
#define LDS_SYN1  269312
#define LDS_MEM1  277504
#define LDS_SYN2  285696
#define LDS_MEM2  293888
#define LDS_B1    302080
#define LDS_B2    304128
#define LDS_SCL   306176
#define LDS_SHF   306432
#define LDS_STG   306688
#define LDS_TOTAL 310784

__device__ __forceinline__ v8f_t wmma_fp8_mac(const unsigned char* A, int astride, int kbase,
                                              const unsigned char* W, int wstride, int n0,
                                              int lane, v8f_t c) {
    const int m   = lane & 15;
    const int hi8 = (lane >> 4) << 4;
    const unsigned char* ar = A + m * astride + kbase + hi8;
    uint2 a0 = *(const uint2*)(ar + 0);
    uint2 a1 = *(const uint2*)(ar + 8);
    uint2 a2 = *(const uint2*)(ar + 32);
    uint2 a3 = *(const uint2*)(ar + 40);
    v8i_t av;
    av[0] = (int)a0.x; av[1] = (int)a0.y; av[2] = (int)a1.x; av[3] = (int)a1.y;
    av[4] = (int)a2.x; av[5] = (int)a2.y; av[6] = (int)a3.x; av[7] = (int)a3.y;

    const int n   = n0 + (lane & 15);
    const int kb2 = kbase + ((lane >> 4) << 4);
    const unsigned char* br = W + n * wstride + kb2;
    uint4 b0 = *(const uint4*)(br + 0);
    uint4 b1 = *(const uint4*)(br + 32);
    v8i_t bv;
    bv[0] = (int)b0.x; bv[1] = (int)b0.y; bv[2] = (int)b0.z; bv[3] = (int)b0.w;
    bv[4] = (int)b1.x; bv[5] = (int)b1.y; bv[6] = (int)b1.z; bv[7] = (int)b1.w;

    return __builtin_amdgcn_wmma_f32_16x16x64_fp8_fp8(av, bv, (short)0, c, false, false);
}

__global__ __launch_bounds__(512, 1) void slstm_scan_kernel(
        const __hip_bfloat16* __restrict__ convb,
        const unsigned char* __restrict__ gW1,
        const unsigned char* __restrict__ gW2,
        const float* __restrict__ gB1, const float* __restrict__ gB2,
        const float* __restrict__ gScl, const float* __restrict__ gShf,
        const float* __restrict__ thr1p, const float* __restrict__ thr2p,
        float* __restrict__ features) {
    extern __shared__ unsigned char smem[];
    unsigned char* W1 = smem + LDS_W1;
    unsigned char* W2 = smem + LDS_W2;
    unsigned char* A1 = smem + LDS_A1;
    unsigned char* A2 = smem + LDS_A2;
    float* gates = (float*)(smem + LDS_GATES);
    float* syn1  = (float*)(smem + LDS_SYN1);
    float* mem1  = (float*)(smem + LDS_MEM1);
    float* syn2  = (float*)(smem + LDS_SYN2);
    float* mem2  = (float*)(smem + LDS_MEM2);
    float* bias1 = (float*)(smem + LDS_B1);
    float* bias2 = (float*)(smem + LDS_B2);
    float* scl   = (float*)(smem + LDS_SCL);
    float* shf   = (float*)(smem + LDS_SHF);
    unsigned char* stage = smem + LDS_STG;

    const int tid   = threadIdx.x;
    const int wave  = tid >> 5;
    const int lane  = tid & 31;
    const int brow0 = blockIdx.x * 16;
    const float thr1 = thr1p[0];
    const float thr2 = thr2p[0];

    const int cid  = tid << 1;
    const int cm   = cid >> 6;
    const int cch  = cid & 63;

    {
        const int4* s1 = (const int4*)gW1; int4* d1 = (int4*)W1;
        for (int i = tid; i < (G4 * K1) / 16; i += 512) d1[i] = s1[i];
        const int4* s2 = (const int4*)gW2; int4* d2 = (int4*)W2;
        for (int i = tid; i < (G4 * K2) / 16; i += 512) d2[i] = s2[i];
        if (tid < 512) { bias1[tid] = gB1[tid]; bias2[tid] = gB2[tid]; }
        if (tid < 64)  { scl[tid] = gScl[tid]; shf[tid] = gShf[tid]; }
        for (int i = tid; i < 16 * HID; i += 512) { syn1[i] = 0.f; mem1[i] = 0.f; syn2[i] = 0.f; mem2[i] = 0.f; }
        for (int i = tid; i < 16 * K1; i += 512) A1[i] = 0;
        for (int i = tid; i < 16 * K2; i += 512) A2[i] = 0;
    }
    float4 facc = {0.f, 0.f, 0.f, 0.f};

    {
        unsigned lds0 = (unsigned)(unsigned long long)(stage + (tid << 2));
        const __hip_bfloat16* g0 = convb + (long long)brow0 * CH + cid;
        asm volatile("global_load_async_to_lds_b32 %0, %1, off" :: "v"(lds0), "v"(g0) : "memory");
        unsigned lds1 = (unsigned)(unsigned long long)(stage + 2048 + (tid << 2));
        const __hip_bfloat16* g1 = convb + ((long long)BATCH + brow0) * CH + cid;
        asm volatile("global_load_async_to_lds_b32 %0, %1, off" :: "v"(lds1), "v"(g1) : "memory");
        asm volatile("s_wait_asynccnt 0x1" ::: "memory");
        const __hip_bfloat16* st = (const __hip_bfloat16*)stage + cid;
        float c0 = __bfloat162float(st[0]);
        float c1 = __bfloat162float(st[1]);
        unsigned short sp = (unsigned short)(((c0 * scl[cch] + shf[cch] > 0.0f) ? 0x38u : 0u)
                          | (((c1 * scl[cch + 1] + shf[cch + 1] > 0.0f) ? 0x38u : 0u) << 8));
        *(unsigned short*)(A1 + cm * K1 + cch) = sp;
    }
    __syncthreads();

    for (int t = 0; t < T_LEN; ++t) {
#pragma unroll
        for (int tt = 0; tt < 2; ++tt) {
            int n0 = (wave * 2 + tt) * 16;
            float bv = bias1[n0 + (lane & 15)];
            v8f_t c;
#pragma unroll
            for (int r = 0; r < 8; ++r) c[r] = bv;
#pragma unroll
            for (int kb = 0; kb < K1; kb += 64)
                c = wmma_fp8_mac(A1, K1, kb, W1, K1, n0, lane, c);
            int mrow = (lane >> 4) << 3;
            int ncol = n0 + (lane & 15);
#pragma unroll
            for (int r = 0; r < 8; ++r) gates[(mrow + r) * G4 + ncol] = c[r];
        }
        __syncthreads();

        {
            const int id0 = tid << 2;
            const int m = id0 >> 7, h0 = id0 & 127;
            float4 gi4 = *(const float4*)&gates[m * G4 + h0];
            float4 gf4 = *(const float4*)&gates[m * G4 + 128 + h0];
            float4 gg4 = *(const float4*)&gates[m * G4 + 256 + h0];
            float4 go4 = *(const float4*)&gates[m * G4 + 384 + h0];
            float4 sy  = *(const float4*)&syn1[m * HID + h0];
            float4 me  = *(const float4*)&mem1[m * HID + h0];
            float gi[4] = {gi4.x, gi4.y, gi4.z, gi4.w};
            float gf[4] = {gf4.x, gf4.y, gf4.z, gf4.w};
            float gg[4] = {gg4.x, gg4.y, gg4.z, gg4.w};
            float go[4] = {go4.x, go4.y, go4.z, go4.w};
            float sp[4] = {sy.x, sy.y, sy.z, sy.w};
            float mp[4] = {me.x, me.y, me.z, me.w};
            float cn[4], mn[4];
            unsigned packm = 0, packs = 0;
#pragma unroll
            for (int q = 0; q < 4; ++q) {
                float reset = (mp[q] - thr1 > 0.0f) ? thr1 : 0.0f;
                cn[q] = sigm(gf[q]) * sp[q] + sigm(gi[q]) * fast_tanh(gg[q]);
                float hn = sigm(go[q]) * fast_tanh(cn[q]);
                mn[q] = hn - reset;
                packm |= f32_to_e4m3(mn[q]) << (8 * q);
                packs |= ((mn[q] - thr1 > 0.0f) ? 0x38u : 0u) << (8 * q);
            }
            *(float4*)&syn1[m * HID + h0] = make_float4(cn[0], cn[1], cn[2], cn[3]);
            *(float4*)&mem1[m * HID + h0] = make_float4(mn[0], mn[1], mn[2], mn[3]);
            *(unsigned*)(A1 + m * K1 + CH + h0) = packm;
            *(unsigned*)(A2 + m * K2 + h0)      = packs;
        }
        __syncthreads();

#pragma unroll
        for (int tt = 0; tt < 2; ++tt) {
            int n0 = (wave * 2 + tt) * 16;
            float bv = bias2[n0 + (lane & 15)];
            v8f_t c;
#pragma unroll
            for (int r = 0; r < 8; ++r) c[r] = bv;
#pragma unroll
            for (int kb = 0; kb < K2; kb += 64)
                c = wmma_fp8_mac(A2, K2, kb, W2, K2, n0, lane, c);
            int mrow = (lane >> 4) << 3;
            int ncol = n0 + (lane & 15);
#pragma unroll
            for (int r = 0; r < 8; ++r) gates[(mrow + r) * G4 + ncol] = c[r];
        }
        __syncthreads();

        {
            const int id0 = tid << 2;
            const int m = id0 >> 7, h0 = id0 & 127;
            float4 gi4 = *(const float4*)&gates[m * G4 + h0];
            float4 gf4 = *(const float4*)&gates[m * G4 + 128 + h0];
            float4 gg4 = *(const float4*)&gates[m * G4 + 256 + h0];
            float4 go4 = *(const float4*)&gates[m * G4 + 384 + h0];
            float4 sy  = *(const float4*)&syn2[m * HID + h0];
            float4 me  = *(const float4*)&mem2[m * HID + h0];
            float gi[4] = {gi4.x, gi4.y, gi4.z, gi4.w};
            float gf[4] = {gf4.x, gf4.y, gf4.z, gf4.w};
            float gg[4] = {gg4.x, gg4.y, gg4.z, gg4.w};
            float go[4] = {go4.x, go4.y, go4.z, go4.w};
            float sp[4] = {sy.x, sy.y, sy.z, sy.w};
            float mp[4] = {me.x, me.y, me.z, me.w};
            float cn[4], mn[4];
            unsigned packm = 0;
#pragma unroll
            for (int q = 0; q < 4; ++q) {
                float reset = (mp[q] - thr2 > 0.0f) ? thr2 : 0.0f;
                cn[q] = sigm(gf[q]) * sp[q] + sigm(gi[q]) * fast_tanh(gg[q]);
                float hn = sigm(go[q]) * fast_tanh(cn[q]);
                mn[q] = hn - reset;
                packm |= f32_to_e4m3(mn[q]) << (8 * q);
            }
            *(float4*)&syn2[m * HID + h0] = make_float4(cn[0], cn[1], cn[2], cn[3]);
            *(float4*)&mem2[m * HID + h0] = make_float4(mn[0], mn[1], mn[2], mn[3]);
            *(unsigned*)(A2 + m * K2 + HID + h0) = packm;
            facc.x += mn[0]; facc.y += mn[1]; facc.z += mn[2]; facc.w += mn[3];
        }
        if (t + 1 < T_LEN) {
            if (t + 2 < T_LEN) {
                unsigned ldso = (unsigned)(unsigned long long)(stage + ((t + 2) & 1) * 2048 + (tid << 2));
                const __hip_bfloat16* gp = convb + ((long long)(t + 2) * BATCH + brow0) * CH + cid;
                asm volatile("global_load_async_to_lds_b32 %0, %1, off" :: "v"(ldso), "v"(gp) : "memory");
                asm volatile("s_wait_asynccnt 0x1" ::: "memory");
            } else {
                asm volatile("s_wait_asynccnt 0x0" ::: "memory");
            }
            const __hip_bfloat16* st = (const __hip_bfloat16*)(stage + ((t + 1) & 1) * 2048) + cid;
            float c0 = __bfloat162float(st[0]);
            float c1 = __bfloat162float(st[1]);
            unsigned short sp = (unsigned short)(((c0 * scl[cch] + shf[cch] > 0.0f) ? 0x38u : 0u)
                              | (((c1 * scl[cch + 1] + shf[cch + 1] > 0.0f) ? 0x38u : 0u) << 8));
            *(unsigned short*)(A1 + cm * K1 + cch) = sp;
        }
        __syncthreads();
    }

    {
        const int id0 = tid << 2;
        const int m = id0 >> 7, h0 = id0 & 127;
        const float inv = 1.0f / (float)T_LEN;
        *(float4*)&features[(brow0 + m) * HID + h0] =
            make_float4(facc.x * inv, facc.y * inv, facc.z * inv, facc.w * inv);
    }
}

__global__ __launch_bounds__(64) void head_kernel(const float* __restrict__ feat,
                                                  const float* __restrict__ gw, const float* __restrict__ gb,
                                                  const float* __restrict__ d1w, const float* __restrict__ d1b,
                                                  const float* __restrict__ thr_dom,
                                                  const float* __restrict__ d2w, const float* __restrict__ d2b,
                                                  float* __restrict__ out) {
    __shared__ float f[HID];
    __shared__ float dh[64];
    const int b = blockIdx.x;
    for (int i = threadIdx.x; i < HID; i += 64) f[i] = feat[b * HID + i];
    __syncthreads();
    const int tid = threadIdx.x;
    {
        float a = d1b[tid];
        for (int h = 0; h < HID; ++h) a += f[h] * d1w[tid * HID + h];
        dh[tid] = (a - thr_dom[0] > 0.0f) ? 1.0f : 0.0f;
    }
    __syncthreads();
    if (tid < 8) {
        float a = gb[tid];
        for (int h = 0; h < HID; ++h) a += f[h] * gw[tid * HID + h];
        out[b * 8 + tid] = a;
    }
    if (tid < 10) {
        float a = d2b[tid];
        for (int k = 0; k < 64; ++k) a += dh[k] * d2w[tid * 64 + k];
        out[BATCH * 8 + b * 10 + tid] = a;
    }
}

#define WS_CONV   ((size_t)0)
#define WS_STATS  ((size_t)65536000)
#define WS_SCALE  ((size_t)65536512)
#define WS_SHIFT  ((size_t)65536768)
#define WS_W1     ((size_t)65537024)
#define WS_W2     ((size_t)65635328)
#define WS_BIAS1  ((size_t)65766400)
#define WS_BIAS2  ((size_t)65768448)
#define WS_FEAT   ((size_t)65770496)

extern "C" void kernel_launch(void* const* d_in, const int* in_sizes, int n_in,
                              void* d_out, int out_size, void* d_ws, size_t ws_size,
                              hipStream_t stream) {
    (void)in_sizes; (void)n_in; (void)out_size; (void)ws_size;
    const float* x        = (const float*)d_in[0];
    const float* conv_w   = (const float*)d_in[1];
    const float* bn_gamma = (const float*)d_in[2];
    const float* bn_beta  = (const float*)d_in[3];
    const float* thr_lif1 = (const float*)d_in[4];
    const float* w_ih1    = (const float*)d_in[5];
    const float* w_hh1    = (const float*)d_in[6];
    const float* b_ih1    = (const float*)d_in[7];
    const float* b_hh1    = (const float*)d_in[8];
    const float* thr1     = (const float*)d_in[9];
    const float* w_ih2    = (const float*)d_in[10];
    const float* w_hh2    = (const float*)d_in[11];
    const float* b_ih2    = (const float*)d_in[12];
    const float* b_hh2    = (const float*)d_in[13];
    const float* thr2     = (const float*)d_in[14];
    const float* fc_g_w   = (const float*)d_in[15];
    const float* fc_g_b   = (const float*)d_in[16];
    const float* fc_d1_w  = (const float*)d_in[17];
    const float* fc_d1_b  = (const float*)d_in[18];
    const float* thr_dom  = (const float*)d_in[19];
    const float* fc_d2_w  = (const float*)d_in[20];
    const float* fc_d2_b  = (const float*)d_in[21];
    float* out = (float*)d_out;

    unsigned char* ws = (unsigned char*)d_ws;
    __hip_bfloat16* convb = (__hip_bfloat16*)(ws + WS_CONV);
    float* stats  = (float*)(ws + WS_STATS);
    float* scale  = (float*)(ws + WS_SCALE);
    float* shiftT = (float*)(ws + WS_SHIFT);
    unsigned char* W1 = ws + WS_W1;
    unsigned char* W2 = ws + WS_W2;
    float* bias1 = (float*)(ws + WS_BIAS1);
    float* bias2 = (float*)(ws + WS_BIAS2);
    float* feat  = (float*)(ws + WS_FEAT);

    zero_stats_kernel<<<1, 128, 0, stream>>>(stats);
    conv_stats_kernel<<<4096, 256, 0, stream>>>(x, conv_w, convb, stats);
    bn_finalize_kernel<<<1, 64, 0, stream>>>(stats, bn_gamma, bn_beta, thr_lif1, scale, shiftT);
    pack_kernel<<<64, 256, 0, stream>>>(w_ih1, w_hh1, b_ih1, b_hh1,
                                        w_ih2, w_hh2, b_ih2, b_hh2,
                                        W1, W2, bias1, bias2);

    (void)hipFuncSetAttribute((const void*)slstm_scan_kernel,
                              hipFuncAttributeMaxDynamicSharedMemorySize, LDS_TOTAL);
    slstm_scan_kernel<<<16, 512, LDS_TOTAL, stream>>>(convb, W1, W2, bias1, bias2,
                                                      scale, shiftT, thr1, thr2, feat);

    head_kernel<<<BATCH, 64, 0, stream>>>(feat, fc_g_w, fc_g_b, fc_d1_w, fc_d1_b,
                                          thr_dom, fc_d2_w, fc_d2_b, out);
}
